// EquShiftQ2DF_68281390072660
// MI455X (gfx1250) — compile-verified, matrix operand order checked by reading, not run on hardware
//
#include <hip/hip_runtime.h>
#include <hip/hip_bf16.h>
#include <stdint.h>

typedef __bf16 bf16_t;
typedef __attribute__((ext_vector_type(16))) __bf16 v16bf;
typedef __attribute__((ext_vector_type(8)))  float  v8f;

#define WMMA_BF16(a,b,c) \
  __builtin_amdgcn_wmma_f32_16x16x32_bf16(false,(a),false,(b),(short)0,(c),false,false)

constexpr int SBUF = 128 * 40;   // one staging buffer in halves (padded rows)

__device__ __forceinline__ void copy32(bf16_t* dst, const bf16_t* src) {
  ((uint4*)dst)[0] = ((const uint4*)src)[0];
  ((uint4*)dst)[1] = ((const uint4*)src)[1];
  ((uint4*)dst)[2] = ((const uint4*)src)[2];
  ((uint4*)dst)[3] = ((const uint4*)src)[3];
}
__device__ __forceinline__ void zero32(bf16_t* dst) {
  uint4 z; z.x = 0u; z.y = 0u; z.z = 0u; z.w = 0u;
  ((uint4*)dst)[0] = z; ((uint4*)dst)[1] = z;
  ((uint4*)dst)[2] = z; ((uint4*)dst)[3] = z;
}

// CDNA5 async global -> LDS copy (ASYNCcnt), 16B per lane per issue.
__device__ __forceinline__ void async_b128(const bf16_t* lds, const bf16_t* g) {
  uint32_t loff = (uint32_t)(uintptr_t)lds;           // low 32 bits = LDS offset
  uint64_t ga   = (uint64_t)(uintptr_t)g;
  asm volatile("global_load_async_to_lds_b128 %0, %1, off"
               :: "v"(loff), "v"(ga) : "memory");
}
__device__ __forceinline__ void async_row(const bf16_t* lds, const bf16_t* g) {
  async_b128(lds,      g);
  async_b128(lds + 8,  g + 8);
  async_b128(lds + 16, g + 16);
  async_b128(lds + 24, g + 24);
}
// async loads retire in order: <=8 outstanding means the older stage group landed
__device__ __forceinline__ void wait_async0() { asm volatile("s_wait_asynccnt 0x0" ::: "memory"); }
__device__ __forceinline__ void wait_async8() { asm volatile("s_wait_asynccnt 0x8" ::: "memory"); }

// ---------------------------------------------------------------------------
// bf16 GEMM:  C = act( A(M,K) * W(N,K)^T + bias );  K % 32 == 0 required.
// 128 threads (4 wave32), tile 128x128, BK=32, 64x64 per wave (16 WMMA).
// ASYNC: M%128==0 && N%128==0; double-buffered async global->LDS staging.
// SPLITK: gridDim.z K-slices of Kc each; f32 partials, no bias/act.
// ---------------------------------------------------------------------------
template<bool RELU, bool OUT_BF16, bool ASYNC, bool SPLITK>
__global__ __launch_bounds__(128) void k_gemm(
    const bf16_t* __restrict__ A, const bf16_t* __restrict__ W,
    const float* __restrict__ bias, void* __restrict__ Cout,
    int M, int N, int K, int ldc, int Kc)
{
  __shared__ bf16_t sA[(ASYNC ? 2 : 1) * SBUF];
  __shared__ bf16_t sB[(ASYNC ? 2 : 1) * SBUF];
  const int tid  = threadIdx.x;
  const int lane = tid & 31;
  const int wave = tid >> 5;
  const int bm = blockIdx.y * 128;
  const int bn = blockIdx.x * 128;
  const int wm = (wave >> 1) * 64;
  const int wn = (wave & 1) * 64;
  v8f acc[4][4] = {};

  int kbeg = 0, kend = K;
  if (SPLITK) { kbeg = blockIdx.z * Kc; kend = kbeg + Kc; }
  const int nk = (kend - kbeg) >> 5;

  const int gr = bm + tid;
  const int gn = bn + tid;
  const bf16_t* arow = A + (size_t)gr * K;
  const bf16_t* brow = W + (size_t)gn * K;

  if (ASYNC) {
    async_row(&sA[tid * 40], arow + kbeg);
    async_row(&sB[tid * 40], brow + kbeg);
  }

  for (int ik = 0; ik < nk; ++ik) {
    const int cur = ASYNC ? (ik & 1) : 0;
    if (ASYNC) {
      if (ik + 1 < nk) {
        const int nb = (ik + 1) & 1;
        const int kn = kbeg + ((ik + 1) << 5);
        async_row(&sA[nb * SBUF + tid * 40], arow + kn);
        async_row(&sB[nb * SBUF + tid * 40], brow + kn);
        wait_async8();
      } else {
        wait_async0();
      }
    } else {
      const int k0 = kbeg + (ik << 5);
      if (gr < M) copy32(&sA[tid * 40], arow + k0); else zero32(&sA[tid * 40]);
      if (gn < N) copy32(&sB[tid * 40], brow + k0); else zero32(&sB[tid * 40]);
    }
    __syncthreads();
    const bf16_t* cA = &sA[cur * SBUF];
    const bf16_t* cB = &sB[cur * SBUF];
    const int fr = lane & 15;
    const int kh = (lane >> 4) * 8;
    v16bf fa[4], fb[4];
#pragma unroll
    for (int i = 0; i < 4; ++i) {
      const bf16_t* pa = &cA[(wm + i * 16 + fr) * 40 + kh];
      ((uint4*)&fa[i])[0] = *(const uint4*)pa;
      ((uint4*)&fa[i])[1] = *(const uint4*)(pa + 16);
      const bf16_t* pb = &cB[(wn + i * 16 + fr) * 40 + kh];
      ((uint4*)&fb[i])[0] = *(const uint4*)pb;
      ((uint4*)&fb[i])[1] = *(const uint4*)(pb + 16);
    }
#pragma unroll
    for (int i = 0; i < 4; ++i)
#pragma unroll
      for (int j = 0; j < 4; ++j)
        acc[i][j] = WMMA_BF16(fa[i], fb[j], acc[i][j]);
    __syncthreads();
  }

  const int cn  = lane & 15;
  const int rhi = (lane >> 4) * 8;
#pragma unroll
  for (int i = 0; i < 4; ++i) {
#pragma unroll
    for (int j = 0; j < 4; ++j) {
      int col = bn + wn + j * 16 + cn;
      if (ASYNC || col < N) {
        float bv = SPLITK ? 0.0f : bias[col];
#pragma unroll
        for (int r = 0; r < 8; ++r) {
          int row = bm + wm + i * 16 + rhi + r;
          if (ASYNC || row < M) {
            float v = acc[i][j][r] + bv;
            if (RELU) v = v > 0.0f ? v : 0.0f;
            if (SPLITK)
              ((float*)Cout)[(size_t)blockIdx.z * M * N + (size_t)row * N + col] = v;
            else if (OUT_BF16)
              ((bf16_t*)Cout)[(size_t)row * ldc + col] = (bf16_t)v;
            else
              ((float*)Cout)[(size_t)row * ldc + col] = v;
          }
        }
      }
    }
  }
}

// ---------------------------------------------------------------------------
// Implicit-GEMM 3x3 conv, NHWC bf16 in, W = (O, ky*3+kx, C) bf16, NHWC bf16 out.
// M = B*OH*OW, N = O, K = 9*C.  ASYNC: pad==0, C>=32, M%128==0, N%128==0.
// SPLITK: f32 partials per gridDim.z slice, no bias/act.
// ---------------------------------------------------------------------------
template<bool RELU, bool ASYNC, bool SPLITK>
__global__ __launch_bounds__(128) void k_conv(
    const bf16_t* __restrict__ X, const bf16_t* __restrict__ W,
    const float* __restrict__ bias, void* __restrict__ Yout,
    int M, int N, int K, int IH, int IW, int OH, int OW,
    int cshift, int stride, int pad, int Kc)
{
  __shared__ bf16_t sA[(ASYNC ? 2 : 1) * SBUF];
  __shared__ bf16_t sB[(ASYNC ? 2 : 1) * SBUF];
  const int C = 1 << cshift;
  const int tid  = threadIdx.x;
  const int lane = tid & 31;
  const int wave = tid >> 5;
  const int bm = blockIdx.y * 128;
  const int bn = blockIdx.x * 128;
  const int wm = (wave >> 1) * 64;
  const int wn = (wave & 1) * 64;
  const int OHW = OH * OW;
  v8f acc[4][4] = {};

  int kbeg = 0, kend = K;
  if (SPLITK) { kbeg = blockIdx.z * Kc; kend = kbeg + Kc; }
  const int nk = (kend - kbeg) >> 5;

  const int gr = bm + tid;
  const bool rowok = gr < M;
  int pb = 0, poy = 0, pox = 0;
  if (rowok) {
    pb = gr / OHW;
    int r2 = gr - pb * OHW;
    poy = r2 / OW;
    pox = r2 - poy * OW;
  }
  const int gn = bn + tid;
  const bf16_t* brow = W + (size_t)gn * K;

  // gather address for a 32-run starting at k (same (ky,kx) when C%32==0, pad==0)
  auto asrc = [&](int k) -> const bf16_t* {
    int kyx = k >> cshift;
    int c0  = k & (C - 1);
    int ky = kyx / 3, kx = kyx - ky * 3;
    int iy = poy * stride + ky;
    int ix = pox * stride + kx;
    return X + ((size_t)((pb * IH + iy) * IW + ix) << cshift) + c0;
  };

  if (ASYNC) {
    async_row(&sA[tid * 40], asrc(kbeg));
    async_row(&sB[tid * 40], brow + kbeg);
  }

  for (int ik = 0; ik < nk; ++ik) {
    const int cur = ASYNC ? (ik & 1) : 0;
    if (ASYNC) {
      if (ik + 1 < nk) {
        const int nb = (ik + 1) & 1;
        const int kn = kbeg + ((ik + 1) << 5);
        async_row(&sA[nb * SBUF + tid * 40], asrc(kn));
        async_row(&sB[nb * SBUF + tid * 40], brow + kn);
        wait_async8();
      } else {
        wait_async0();
      }
    } else {
      const int k0 = kbeg + (ik << 5);
      bf16_t* sAr = &sA[tid * 40];
      if (rowok) {
        if (cshift >= 5) {
          int kyx = k0 >> cshift;
          int c0  = k0 & (C - 1);
          int ky = kyx / 3, kx = kyx - ky * 3;
          int iy = poy * stride - pad + ky;
          int ix = pox * stride - pad + kx;
          if ((unsigned)iy < (unsigned)IH && (unsigned)ix < (unsigned)IW)
            copy32(sAr, X + ((size_t)((pb * IH + iy) * IW + ix) << cshift) + c0);
          else
            zero32(sAr);
        } else {
#pragma unroll 4
          for (int j = 0; j < 32; ++j) {
            int k = k0 + j;
            bf16_t v = (bf16_t)0.0f;
            if (k < K) {
              int kyx = k >> cshift;
              int c   = k & (C - 1);
              int ky = kyx / 3, kx = kyx - ky * 3;
              int iy = poy * stride - pad + ky;
              int ix = pox * stride - pad + kx;
              if ((unsigned)iy < (unsigned)IH && (unsigned)ix < (unsigned)IW)
                v = X[((size_t)((pb * IH + iy) * IW + ix) << cshift) + c];
            }
            sAr[j] = v;
          }
        }
      } else zero32(sAr);
      if (gn < N) copy32(&sB[tid * 40], brow + k0); else zero32(&sB[tid * 40]);
    }
    __syncthreads();
    const bf16_t* cA = &sA[cur * SBUF];
    const bf16_t* cB = &sB[cur * SBUF];
    const int fr = lane & 15;
    const int kh = (lane >> 4) * 8;
    v16bf fa[4], fb[4];
#pragma unroll
    for (int i = 0; i < 4; ++i) {
      const bf16_t* pa = &cA[(wm + i * 16 + fr) * 40 + kh];
      ((uint4*)&fa[i])[0] = *(const uint4*)pa;
      ((uint4*)&fa[i])[1] = *(const uint4*)(pa + 16);
      const bf16_t* pbp = &cB[(wn + i * 16 + fr) * 40 + kh];
      ((uint4*)&fb[i])[0] = *(const uint4*)pbp;
      ((uint4*)&fb[i])[1] = *(const uint4*)(pbp + 16);
    }
#pragma unroll
    for (int i = 0; i < 4; ++i)
#pragma unroll
      for (int j = 0; j < 4; ++j)
        acc[i][j] = WMMA_BF16(fa[i], fb[j], acc[i][j]);
    __syncthreads();
  }

  const int cn  = lane & 15;
  const int rhi = (lane >> 4) * 8;
#pragma unroll
  for (int i = 0; i < 4; ++i) {
#pragma unroll
    for (int j = 0; j < 4; ++j) {
      int col = bn + wn + j * 16 + cn;
      if (ASYNC || col < N) {
        float bv = SPLITK ? 0.0f : bias[col];
#pragma unroll
        for (int r = 0; r < 8; ++r) {
          int row = bm + wm + i * 16 + rhi + r;
          if (ASYNC || row < M) {
            float v = acc[i][j][r] + bv;
            if (RELU) v = v > 0.0f ? v : 0.0f;
            if (SPLITK)
              ((float*)Yout)[(size_t)blockIdx.z * M * N + (size_t)row * N + col] = v;
            else
              ((bf16_t*)Yout)[(size_t)row * N + col] = (bf16_t)v;
          }
        }
      }
    }
  }
}

// ---------------------------------------------------------------------------
// Split-K reduction: fixed-order sum of S partials + bias + ReLU -> bf16
// ---------------------------------------------------------------------------
__global__ void k_redux(const float* __restrict__ part, const float* __restrict__ bias,
                        bf16_t* __restrict__ out, int M, int N, int ldc, int S) {
  int idx = blockIdx.x * 256 + threadIdx.x;
  if (idx >= M * N) return;
  int row = idx / N, col = idx - row * N;
  float a = bias[col];
  for (int s = 0; s < S; ++s) a += part[(size_t)s * M * N + idx];
  a = a > 0.0f ? a : 0.0f;
  out[(size_t)row * ldc + col] = (bf16_t)a;
}

// ---------------------------------------------------------------------------
// Small utility kernels
// ---------------------------------------------------------------------------
__global__ void k_cvt_flat(const float* __restrict__ src, bf16_t* __restrict__ dst, size_t n) {
  size_t i = (size_t)blockIdx.x * blockDim.x + threadIdx.x;
  if (i < n) dst[i] = (bf16_t)src[i];
}

__global__ void k_cvt_ohwi(const float* __restrict__ src, bf16_t* __restrict__ dst,
                           int O, int I, int KHW) {
  size_t n = (size_t)O * I * KHW;
  size_t idx = (size_t)blockIdx.x * blockDim.x + threadIdx.x;
  if (idx >= n) return;
  int o = (int)(idx / ((size_t)I * KHW));
  int rem = (int)(idx - (size_t)o * I * KHW);
  int p = rem / I;
  int i = rem - p * I;
  dst[idx] = (bf16_t)src[((size_t)o * I + i) * KHW + p];
}

__global__ void k_split_patch(const float* __restrict__ patch,
                              bf16_t* __restrict__ img, bf16_t* __restrict__ inh, int B) {
  size_t n = (size_t)B * 576;
  size_t idx = (size_t)blockIdx.x * blockDim.x + threadIdx.x;
  if (idx >= n) return;
  int b = (int)(idx / 576);
  int p = (int)(idx - (size_t)b * 576);
  img[idx] = (bf16_t)patch[(size_t)b * 1152 + p];
  inh[idx] = (bf16_t)patch[(size_t)b * 1152 + 576 + p];
}

__global__ void k_maxpool2(const bf16_t* __restrict__ in, bf16_t* __restrict__ out,
                           int B, int H, int W, int C) {
  int OHh = H >> 1, OWw = W >> 1;
  size_t n = (size_t)B * OHh * OWw * C;
  size_t idx = (size_t)blockIdx.x * blockDim.x + threadIdx.x;
  if (idx >= n) return;
  int c = (int)(idx % C);
  size_t r = idx / C;
  int ox = (int)(r % OWw); r /= OWw;
  int oy = (int)(r % OHh);
  int b  = (int)(r / OHh);
  size_t base = ((size_t)(b * H + oy * 2) * W + ox * 2) * C + c;
  float v0 = (float)in[base];
  float v1 = (float)in[base + C];
  float v2 = (float)in[base + (size_t)W * C];
  float v3 = (float)in[base + (size_t)W * C + C];
  float m = v0 > v1 ? v0 : v1;
  float m2 = v2 > v3 ? v2 : v3;
  out[idx] = (bf16_t)(m > m2 ? m : m2);
}

// ---------------------------------------------------------------------------
// Fused head: dynamic-filter contraction + bias/relu + final FC
// ---------------------------------------------------------------------------
__global__ __launch_bounds__(256) void k_head(
    const float* __restrict__ wb, const float* __restrict__ basis,
    const bf16_t* __restrict__ h, const float* __restrict__ c2w,
    const float* __restrict__ c2b, float* __restrict__ out)
{
  __shared__ float sco[2048];
  __shared__ float sh[576];
  __shared__ float sS[2048];
  __shared__ float sfeat[64];
  int b = blockIdx.x, t = threadIdx.x;

  for (int i = t; i < 2048; i += 256) sco[i] = wb[(size_t)b * 2056 + i];
  for (int i = t; i < 576; i += 256) {
    int ch = i / 9, p = i - ch * 9;
    sh[i] = (float)h[((size_t)b * 9 + p) * 64 + ch];
  }
  __syncthreads();

  for (int s = t; s < 2048; s += 256) {
    int c = s >> 6, d = (s >> 3) & 7, g = s & 7;
    const float* bp = basis + (size_t)((c * 8 + d) * 8) * 9;
    float acc = 0.0f;
#pragma unroll
    for (int e = 0; e < 8; ++e) {
      const float* hp = &sh[(g * 8 + e) * 9];
#pragma unroll
      for (int p = 0; p < 9; ++p) acc += bp[e * 9 + p] * hp[p];
    }
    sS[s] = acc;
  }
  __syncthreads();

  if (t < 64) {
    int f = t >> 3, d = t & 7;
    float acc = 0.0f;
#pragma unroll
    for (int g = 0; g < 8; ++g) {
      const float* cp = &sco[f * 256 + g * 32];
      for (int c = 0; c < 32; ++c) acc += cp[c] * sS[c * 64 + d * 8 + g];
    }
    acc += wb[(size_t)b * 2056 + 2048 + f];
    sfeat[t] = acc > 0.0f ? acc : 0.0f;
  }
  __syncthreads();

  if (t < 16) {
    float acc = c2b[t];
    for (int o = 0; o < 64; ++o) acc += sfeat[o] * c2w[t * 64 + o];
    out[(size_t)b * 16 + t] = acc;
  }
}

// ---------------------------------------------------------------------------
static inline unsigned cdiv(unsigned a, unsigned b) { return (a + b - 1) / b; }

extern "C" void kernel_launch(void* const* d_in, const int* in_sizes, int n_in,
                              void* d_out, int out_size, void* d_ws, size_t ws_size,
                              hipStream_t stream) {
  (void)in_sizes; (void)n_in; (void)out_size; (void)ws_size;
  const int B = 128;
  const float* obs     = (const float*)d_in[0];
  const float* patch   = (const float*)d_in[1];
  const float* enc_w1  = (const float*)d_in[2];
  const float* enc_b1  = (const float*)d_in[3];
  const float* enc_w2  = (const float*)d_in[4];
  const float* enc_b2  = (const float*)d_in[5];
  const float* enc_w3  = (const float*)d_in[6];
  const float* enc_b3  = (const float*)d_in[7];
  const float* pc_w    = (const float*)d_in[8];
  const float* pc_b    = (const float*)d_in[9];
  const float* es_w1   = (const float*)d_in[10];
  const float* es_b1   = (const float*)d_in[11];
  const float* es_w2   = (const float*)d_in[12];
  const float* es_b2   = (const float*)d_in[13];
  const float* ih_w1   = (const float*)d_in[14];
  const float* ih_b1   = (const float*)d_in[15];
  const float* ih_w2   = (const float*)d_in[16];
  const float* ih_b2   = (const float*)d_in[17];
  const float* ih_fc_w = (const float*)d_in[18];
  const float* ih_fc_b = (const float*)d_in[19];
  const float* df_w    = (const float*)d_in[20];
  const float* df_b    = (const float*)d_in[21];
  const float* basis   = (const float*)d_in[22];
  const float* c2_w    = (const float*)d_in[23];
  const float* c2_b    = (const float*)d_in[24];
  float* out = (float*)d_out;

  char* wp = (char*)d_ws;
  auto alloc = [&](size_t bytes) -> void* {
    void* r = (void*)wp;
    wp += (bytes + 255) & ~(size_t)255;
    return r;
  };
  bf16_t* obs_b   = (bf16_t*)alloc((size_t)B * 16384 * 2);
  bf16_t* img_b   = (bf16_t*)alloc((size_t)B * 576 * 2);
  bf16_t* inh_b   = (bf16_t*)alloc((size_t)B * 576 * 2);
  bf16_t* es_w1b  = (bf16_t*)alloc((size_t)1024 * 16384 * 2);
  bf16_t* es_w2b  = (bf16_t*)alloc((size_t)512 * 1024 * 2);
  bf16_t* ih_w1b  = (bf16_t*)alloc((size_t)64 * 9 * 2);
  bf16_t* ih_w2b  = (bf16_t*)alloc((size_t)128 * 576 * 2);
  bf16_t* ih_fcwb = (bf16_t*)alloc((size_t)512 * 4608 * 2);
  bf16_t* df_wb   = (bf16_t*)alloc((size_t)2056 * 1024 * 2);
  bf16_t* enc_w1b = (bf16_t*)alloc((size_t)256 * 9 * 2);
  bf16_t* enc_w2b = (bf16_t*)alloc((size_t)512 * 2304 * 2);
  bf16_t* enc_w3b = (bf16_t*)alloc((size_t)1024 * 4608 * 2);
  bf16_t* pc_wb   = (bf16_t*)alloc((size_t)64 * 9216 * 2);
  bf16_t* es1o    = (bf16_t*)alloc((size_t)B * 1024 * 2);
  bf16_t* concat  = (bf16_t*)alloc((size_t)B * 1024 * 2);
  bf16_t* ih1o    = (bf16_t*)alloc((size_t)B * 144 * 64 * 2);
  bf16_t* ih2o    = (bf16_t*)alloc((size_t)B * 36 * 128 * 2);
  float*  wbf     = (float*) alloc((size_t)B * 2056 * 4);
  bf16_t* enc1o   = (bf16_t*)alloc((size_t)B * 484 * 256 * 2);
  bf16_t* enc2o   = (bf16_t*)alloc((size_t)B * 400 * 512 * 2);
  bf16_t* poolo   = (bf16_t*)alloc((size_t)B * 100 * 512 * 2);
  bf16_t* enc3o   = (bf16_t*)alloc((size_t)B * 64 * 1024 * 2);
  bf16_t* pco     = (bf16_t*)alloc((size_t)B * 9 * 64 * 2);
  float*  p_es1   = (float*) alloc((size_t)16 * B * 1024 * 4);  // 16 K-slices
  float*  p_ihfc  = (float*) alloc((size_t)4 * B * 512 * 4);    // 4 K-slices
  float*  p_pc    = (float*) alloc((size_t)8 * 1152 * 64 * 4);  // 8 K-slices

  const dim3 blk128(128), blk256(256);

  // ---- conversions ----
  { size_t n = (size_t)B * 16384;
    k_cvt_flat<<<cdiv((unsigned)n, 256), blk256, 0, stream>>>(obs, obs_b, n); }
  k_split_patch<<<cdiv(B * 576, 256), blk256, 0, stream>>>(patch, img_b, inh_b, B);
  { size_t n = (size_t)1024 * 16384;
    k_cvt_flat<<<cdiv((unsigned)n, 256), blk256, 0, stream>>>(es_w1, es_w1b, n); }
  { size_t n = (size_t)512 * 1024;
    k_cvt_flat<<<cdiv((unsigned)n, 256), blk256, 0, stream>>>(es_w2, es_w2b, n); }
  { size_t n = (size_t)2056 * 1024;
    k_cvt_flat<<<cdiv((unsigned)n, 256), blk256, 0, stream>>>(df_w, df_wb, n); }
  k_cvt_ohwi<<<cdiv(64 * 1 * 9, 256), blk256, 0, stream>>>(ih_w1, ih_w1b, 64, 1, 9);
  k_cvt_ohwi<<<cdiv(128 * 64 * 9, 256), blk256, 0, stream>>>(ih_w2, ih_w2b, 128, 64, 9);
  k_cvt_ohwi<<<cdiv(512 * 128 * 36, 256), blk256, 0, stream>>>(ih_fc_w, ih_fcwb, 512, 128, 36);
  k_cvt_ohwi<<<cdiv(256 * 1 * 9, 256), blk256, 0, stream>>>(enc_w1, enc_w1b, 256, 1, 9);
  k_cvt_ohwi<<<cdiv(512 * 256 * 9, 256), blk256, 0, stream>>>(enc_w2, enc_w2b, 512, 256, 9);
  k_cvt_ohwi<<<cdiv(1024 * 512 * 9, 256), blk256, 0, stream>>>(enc_w3, enc_w3b, 1024, 512, 9);
  k_cvt_ohwi<<<cdiv(64 * 1024 * 9, 256), blk256, 0, stream>>>(pc_w, pc_wb, 64, 1024, 9);

  // ---- es path: split-K (16 slices) + reduction ----
  k_gemm<false, false, true, true><<<dim3(8, 1, 16), blk128, 0, stream>>>(
      obs_b, es_w1b, es_b1, p_es1, B, 1024, 16384, 1024, 1024);
  k_redux<<<cdiv(B * 1024, 256), blk256, 0, stream>>>(p_es1, es_b1, es1o, B, 1024, 1024, 16);
  k_gemm<true, true, true, false><<<dim3(4, 1, 1), blk128, 0, stream>>>(
      es1o, es_w2b, es_b2, concat, B, 512, 1024, 1024, 0);       // cols [0,512)

  // ---- in-hand path ----
  k_conv<true, false, false><<<dim3(1, cdiv(B * 144, 128)), blk128, 0, stream>>>(
      inh_b, ih_w1b, ih_b1, ih1o, B * 144, 64, 9, 24, 24, 12, 12, 0, 2, 1, 0);
  k_conv<true, false, false><<<dim3(1, cdiv(B * 36, 128)), blk128, 0, stream>>>(
      ih1o, ih_w2b, ih_b2, ih2o, B * 36, 128, 576, 12, 12, 6, 6, 6, 2, 1, 0);
  k_gemm<false, false, true, true><<<dim3(4, 1, 4), blk128, 0, stream>>>(
      ih2o, ih_fcwb, ih_fc_b, p_ihfc, B, 512, 4608, 1024, 1152);
  k_redux<<<cdiv(B * 512, 256), blk256, 0, stream>>>(p_ihfc, ih_fc_b, concat + 512, B, 512, 1024, 4);

  // ---- dynamic-filter weights (N=2056 not tile-aligned -> generic path) ----
  k_gemm<false, false, false, false><<<dim3(cdiv(2056, 128), 1), blk128, 0, stream>>>(
      concat, df_wb, df_b, wbf, B, 2056, 1024, 2056, 0);

  // ---- image encoder ----
  k_conv<true, false, false><<<dim3(2, cdiv(B * 484, 128)), blk128, 0, stream>>>(
      img_b, enc_w1b, enc_b1, enc1o, B * 484, 256, 9, 24, 24, 22, 22, 0, 1, 0, 0);
  k_conv<true, true, false><<<dim3(4, cdiv(B * 400, 128)), blk128, 0, stream>>>(
      enc1o, enc_w2b, enc_b2, enc2o, B * 400, 512, 2304, 22, 22, 20, 20, 8, 1, 0, 0);
  { size_t n = (size_t)B * 100 * 512;
    k_maxpool2<<<cdiv((unsigned)n, 256), blk256, 0, stream>>>(enc2o, poolo, B, 20, 20, 512); }
  k_conv<true, true, false><<<dim3(8, cdiv(B * 64, 128)), blk128, 0, stream>>>(
      poolo, enc_w3b, enc_b3, enc3o, B * 64, 1024, 4608, 10, 10, 8, 8, 9, 1, 0, 0);
  k_conv<false, false, true><<<dim3(1, cdiv(B * 9, 128), 8), blk128, 0, stream>>>(
      enc3o, pc_wb, pc_b, p_pc, B * 9, 64, 9216, 8, 8, 3, 3, 10, 2, 0, 1152);
  k_redux<<<cdiv(1152 * 64, 256), blk256, 0, stream>>>(p_pc, pc_b, pco, 1152, 64, 64, 8);

  // ---- head ----
  k_head<<<B, blk256, 0, stream>>>(wbf, basis, pco, c2_w, c2_b, out);
}
